// MultiHeadAttention_1494648619650
// MI455X (gfx1250) — matrix operand order repaired and checked by reading, not compiled, not run on hardware
//
#include <hip/hip_runtime.h>
#include <hip/hip_bf16.h>

// MI455X / gfx1250 multi-head attention forward, WMMA bf16 path + TDM.
// B=2, S=2048, D_MODEL=1024, H=16, DK=64. All dims divisible by tile sizes.

typedef __attribute__((ext_vector_type(16))) __bf16 v16bf;
typedef __attribute__((ext_vector_type(8)))  __bf16 v8bf;
typedef __attribute__((ext_vector_type(8)))  float  v8f;
typedef __attribute__((ext_vector_type(4))) unsigned int u32x4;
typedef __attribute__((ext_vector_type(8))) int          i32x8;
typedef __attribute__((ext_vector_type(4))) int          i32x4;

#define D_MODEL 1024
#define NHEAD   16
#define DK      64
#define BATCH   2
#define SEQ     2048
#define MROWS   (BATCH * SEQ)   // 4096

__device__ __forceinline__ __bf16 cvt_bf16(float f) {
  unsigned u = __builtin_bit_cast(unsigned, f);
  u += 0x7FFFu + ((u >> 16) & 1u);              // round-to-nearest-even
  unsigned short h = (unsigned short)(u >> 16);
  return __builtin_bit_cast(__bf16, h);
}

union FragA { v16bf v; v8bf h[2]; };

// ------------------------------------------------------------------ TDM
// 2D tile load Global -> LDS via Tensor Data Mover.
// D# group0/group1 packed per CDNA5 ISA (CDNA5 ISA).
__device__ __forceinline__ void tdm_load_2d(unsigned lds_byte_addr,
                                            const void* gaddr,
                                            unsigned tile_w,   // elems (2B each)
                                            unsigned tile_h,   // rows
                                            unsigned row_stride_elems,
                                            unsigned tensor_w,
                                            unsigned tensor_h) {
  unsigned long long ga = (unsigned long long)gaddr;
  u32x4 g0;
  g0[0] = 1u;                                   // count=1, user mode
  g0[1] = lds_byte_addr;                        // lds_addr [63:32]
  g0[2] = (unsigned)(ga & 0xFFFFFFFFu);         // global_addr [95:64]
  g0[3] = (unsigned)((ga >> 32) & 0x01FFFFFFu)  // global_addr [120:96]
          | (2u << 30);                         // type=2 ("image") [127:126]
  i32x8 g1;
  g1[0] = (int)(1u << 16);                      // wg_mask=0, data_size=1 (2B)
  g1[1] = (int)((tensor_w & 0xFFFFu) << 16);    // tensor_dim0 [79:48] lo
  g1[2] = (int)(((tensor_w >> 16) & 0xFFFFu) |  // tensor_dim0 hi
                ((tensor_h & 0xFFFFu) << 16));  // tensor_dim1 [111:80] lo
  g1[3] = (int)(((tensor_h >> 16) & 0xFFFFu) |  // tensor_dim1 hi
                ((tile_w & 0xFFFFu) << 16));    // tile_dim0 [127:112]
  g1[4] = (int)(tile_h & 0xFFFFu);              // tile_dim1 [143:128], tile_dim2=0
  g1[5] = (int)row_stride_elems;                // tensor_dim0_stride [207:160] lo
  g1[6] = 0;                                    // stride hi / dim1_stride lo
  g1[7] = 0;                                    // dim1_stride hi
  i32x4 z4 = {0, 0, 0, 0};
#if __has_include(<hip/amd_detail/amd_gfx1250_TDM.h>)
  i32x8 z8 = {};
  __builtin_amdgcn_tensor_load_to_lds(g0, g1, z4, z4, z8, 0);
#else
  __builtin_amdgcn_tensor_load_to_lds(g0, g1, z4, z4, 0);
#endif
}

// ---------------------------------------------------------------- conversions

__global__ void cvt_f32_bf16(const float* __restrict__ in,
                             __bf16* __restrict__ out, int n) {
  int i = blockIdx.x * blockDim.x + threadIdx.x;
  if (i < n) out[i] = cvt_bf16(in[i]);
}

// W: [K][N] f32 row-major  ->  WT: [N][K] bf16 row-major (write-coalesced)
__global__ void transpose_w_bf16(const float* __restrict__ W,
                                 __bf16* __restrict__ WT) {
  int i = blockIdx.x * blockDim.x + threadIdx.x;  // over N*K
  int n = i >> 10;
  int k = i & 1023;
  WT[i] = cvt_bf16(W[k * D_MODEL + n]);
}

// ---------------------------------------------------------------- GEMM
// C[M][N] = A[M][K] @ WT[N][K]^T + bias.  Block: 128x128, 8 waves,
// wave = 32x64 (2 M-tiles x 4 N-tiles), K stepped by 32 (one WMMA K).
// MODE 0: bf16 out [M][N]; MODE 1: f32 out [M][N];
// MODE 2: bf16 out transposed per-head V layout Vt[(b*1024 + col)*SEQ + s].

template <int MODE>
__global__ __launch_bounds__(256)
void gemm_bf16_wmma(const __bf16* __restrict__ A,
                    const __bf16* __restrict__ WT,
                    const float*  __restrict__ bias,
                    void* __restrict__ Cout,
                    int M, int N, int K) {
  const int lane = threadIdx.x & 31;
  const int wave = threadIdx.x >> 5;
  const int half = lane >> 4;          // 0: lanes 0-15, 1: lanes 16-31
  const int l16  = lane & 15;
  const int mblk = blockIdx.y * 128 + (wave & 3) * 32;
  const int nblk = blockIdx.x * 128 + (wave >> 2) * 64;

  v8f acc[2][4];
  v8f zero = {};
#pragma unroll
  for (int mi = 0; mi < 2; ++mi)
#pragma unroll
    for (int ni = 0; ni < 4; ++ni) acc[mi][ni] = zero;

  const int kt_count = K >> 5;
  for (int kt = 0; kt < kt_count; ++kt) {
    const int kb = kt * 32;
    // A fragments: lane l<16 holds row l, K = kb+{0..7,16..23};
    //              lane l>=16 holds row l-16, K = kb+{8..15,24..31}.
    FragA a[2];
#pragma unroll
    for (int mi = 0; mi < 2; ++mi) {
      const __bf16* ap = A + (size_t)(mblk + mi * 16 + l16) * K + kb + half * 8;
      a[mi].h[0] = *(const v8bf*)(ap);
      a[mi].h[1] = *(const v8bf*)(ap + 16);
    }
    // prefetch A one K-tile ahead (global_prefetch_b8)
    __builtin_prefetch(A + (size_t)(mblk + l16) * K + kb + 32, 0, 0);
    // B fragments: lane l<16 holds col l, K = kb+{0..7,16..23};
    //              lane l>=16 holds col l-16, K = kb+{8..15,24..31}.
#pragma unroll
    for (int ni = 0; ni < 4; ++ni) {
      const __bf16* bp =
          WT + (size_t)(nblk + ni * 16 + l16) * K + kb + half * 8;
      FragA bf;
      bf.h[0] = *(const v8bf*)(bp);
      bf.h[1] = *(const v8bf*)(bp + 16);
      v16bf b = bf.v;
#pragma unroll
      for (int mi = 0; mi < 2; ++mi)
        acc[mi][ni] = __builtin_amdgcn_wmma_f32_16x16x32_bf16(
            false, a[mi].v, false, b, (short)0, acc[mi][ni], false, false);
    }
  }

  // Epilogue. C layout: VGPR r -> (row base+half*8+r, col base+l16).
#pragma unroll
  for (int ni = 0; ni < 4; ++ni) {
    const int col = nblk + ni * 16 + l16;
    const float bv = bias[col];
#pragma unroll
    for (int mi = 0; mi < 2; ++mi) {
#pragma unroll
      for (int r = 0; r < 8; ++r) {
        const int row = mblk + mi * 16 + half * 8 + r;
        float val = acc[mi][ni][r] + bv;
        if (MODE == 1) {
          ((float*)Cout)[(size_t)row * N + col] = val;
        } else if (MODE == 0) {
          ((__bf16*)Cout)[(size_t)row * N + col] = cvt_bf16(val);
        } else {
          // V^T head layout: b = row>>11, s = row&2047; index (b*1024+col)*SEQ+s
          const int brow = row >> 11;
          const int srow = row & (SEQ - 1);
          ((__bf16*)Cout)[((size_t)(brow << 10) + col) * SEQ + srow] =
              cvt_bf16(val);
        }
      }
    }
  }
}

// ---------------------------------------------------------------- attention
// Flash-style online softmax. Grid: (S/128, H, B). 8 waves x 16 query rows.
// K block staged into LDS via the Tensor Data Mover; V read from the
// transposed head layout straight from global (32B contiguous B-fragments).

__global__ __launch_bounds__(256)
void flash_attn_wmma(const __bf16* __restrict__ Qh,
                     const __bf16* __restrict__ Kh,
                     const __bf16* __restrict__ Vt,   // [(b*1024+h*64+dk)*SEQ+s]
                     __bf16* __restrict__ Ctx) {
  __shared__ __bf16 sK[128][DK];       // K block (TDM-staged), 16 KB
  __shared__ __bf16 sP[8][16][128];    // per-wave P tile (C->A relayout), 32 KB

  const int lane = threadIdx.x & 31;
  const int wave = threadIdx.x >> 5;
  const int half = lane >> 4;
  const int l16  = lane & 15;
  const int b    = blockIdx.z;
  const int h    = blockIdx.y;
  const int q0   = blockIdx.x * 128;
  const int rowbase = b * SEQ;
  const int hcol = h * DK;

  // Q fragments for this wave's 16x64 tile (2 K-steps of 32), loaded once.
  FragA aq[2];
  {
    const int qrow = rowbase + q0 + wave * 16 + l16;
#pragma unroll
    for (int kt = 0; kt < 2; ++kt) {
      const __bf16* qp =
          Qh + (size_t)qrow * D_MODEL + hcol + kt * 32 + half * 8;
      aq[kt].h[0] = *(const v8bf*)(qp);
      aq[kt].h[1] = *(const v8bf*)(qp + 16);
    }
  }

  v8f o[4];
  v8f zero = {};
#pragma unroll
  for (int nt = 0; nt < 4; ++nt) o[nt] = zero;
  float mrow[8], lrow[8];
#pragma unroll
  for (int r = 0; r < 8; ++r) { mrow[r] = -3.0e38f; lrow[r] = 0.0f; }

  const unsigned sK_lds = (unsigned)(size_t)&sK[0][0];

  for (int j = 0; j < SEQ / 128; ++j) {
    __syncthreads();  // previous iteration done reading sK
    if (wave == 0) {
      // TDM: 128x64 bf16 tile, row stride 1024 elems, into sK[128][64].
      const __bf16* kp = Kh + (size_t)(rowbase + j * 128) * D_MODEL + hcol;
      tdm_load_2d(sK_lds, kp, DK, 128, D_MODEL, D_MODEL, MROWS);
      __builtin_amdgcn_s_wait_tensorcnt(0);
    }
    __syncthreads();  // sK visible to all waves

    // ---- S = (Q K^T) * 1/sqrt(Dk); online row-max over the 16x128 tile.
    float mnew[8];
#pragma unroll
    for (int r = 0; r < 8; ++r) mnew[r] = mrow[r];
    v8f s[8];
#pragma unroll
    for (int nt = 0; nt < 8; ++nt) {
      // B fragment of K^T: lane = key row (two 16B chunks of the sK row).
      const __bf16* kp = &sK[nt * 16 + l16][0];
      FragA fb0, fb1;
      fb0.h[0] = *(const v8bf*)(kp + half * 8);
      fb0.h[1] = *(const v8bf*)(kp + half * 8 + 16);
      fb1.h[0] = *(const v8bf*)(kp + 32 + half * 8);
      fb1.h[1] = *(const v8bf*)(kp + 32 + half * 8 + 16);
      v16bf b0 = fb0.v;        // K-dim 0..31
      v16bf b1 = fb1.v;        // K-dim 32..63
      v8f acc = zero;
      acc = __builtin_amdgcn_wmma_f32_16x16x32_bf16(false, aq[0].v, false, b0,
                                                    (short)0, acc, false, false);
      acc = __builtin_amdgcn_wmma_f32_16x16x32_bf16(false, aq[1].v, false, b1,
                                                    (short)0, acc, false, false);
#pragma unroll
      for (int r = 0; r < 8; ++r) {
        float v = acc[r] * 0.125f;   // 1/sqrt(64)
        acc[r] = v;
        v = fmaxf(v, __shfl_xor(v, 1));
        v = fmaxf(v, __shfl_xor(v, 2));
        v = fmaxf(v, __shfl_xor(v, 4));
        v = fmaxf(v, __shfl_xor(v, 8));  // reduce across 16-lane N group
        mnew[r] = fmaxf(mnew[r], v);
      }
      s[nt] = acc;
    }

    float alpha[8], psum[8];
#pragma unroll
    for (int r = 0; r < 8; ++r) {
      alpha[r] = __expf(mrow[r] - mnew[r]);
      mrow[r]  = mnew[r];
      psum[r]  = 0.0f;
    }

    // P = exp(S - m); stash bf16 into wave-private LDS tile (C -> A layout).
#pragma unroll
    for (int nt = 0; nt < 8; ++nt) {
#pragma unroll
      for (int r = 0; r < 8; ++r) {
        float p = __expf(s[nt][r] - mnew[r]);
        psum[r] += p;
        sP[wave][half * 8 + r][nt * 16 + l16] = cvt_bf16(p);
      }
    }
#pragma unroll
    for (int r = 0; r < 8; ++r) {
      float t = psum[r];
      t += __shfl_xor(t, 1);
      t += __shfl_xor(t, 2);
      t += __shfl_xor(t, 4);
      t += __shfl_xor(t, 8);
      lrow[r] = lrow[r] * alpha[r] + t;
    }
#pragma unroll
    for (int nt = 0; nt < 4; ++nt)
#pragma unroll
      for (int r = 0; r < 8; ++r) o[nt][r] *= alpha[r];

    // ---- O += P @ V  (K-dim = 128 keys, 4 steps of 32; N = Dk = 4 tiles).
    // V B-fragments straight from global transposed layout: 32B contiguous.
#pragma unroll
    for (int kt = 0; kt < 4; ++kt) {
      FragA ap;
      const __bf16* pp = &sP[wave][l16][kt * 32 + half * 16];
      ap.h[0] = *(const v8bf*)(pp);
      ap.h[1] = *(const v8bf*)(pp + 8);
      const int kk = j * 128 + kt * 32 + half * 16;   // key index (16 consec)
#pragma unroll
      for (int nt = 0; nt < 4; ++nt) {
        const int dkcol = hcol + nt * 16 + l16;
        v16bf bv =
            *(const v16bf*)(Vt + ((size_t)(b << 10) + dkcol) * SEQ + kk);
        o[nt] = __builtin_amdgcn_wmma_f32_16x16x32_bf16(
            false, ap.v, false, bv, (short)0, o[nt], false, false);
      }
    }
  }

  // Normalize and emit context in [B*S][D_MODEL] layout (head h columns).
#pragma unroll
  for (int r = 0; r < 8; ++r) {
    const float inv = 1.0f / lrow[r];
    const int row = rowbase + q0 + wave * 16 + half * 8 + r;
#pragma unroll
    for (int nt = 0; nt < 4; ++nt) {
      Ctx[(size_t)row * D_MODEL + hcol + nt * 16 + l16] =
          cvt_bf16(o[nt][r] * inv);
    }
  }
}

// ---------------------------------------------------------------- launch

extern "C" void kernel_launch(void* const* d_in, const int* in_sizes, int n_in,
                              void* d_out, int out_size, void* d_ws,
                              size_t ws_size, hipStream_t stream) {
  const float* q  = (const float*)d_in[0];
  const float* k  = (const float*)d_in[1];
  const float* v  = (const float*)d_in[2];
  const float* Wq = (const float*)d_in[3];
  const float* bq = (const float*)d_in[4];
  const float* Wk = (const float*)d_in[5];
  const float* bk = (const float*)d_in[6];
  const float* Wv = (const float*)d_in[7];
  const float* bv = (const float*)d_in[8];
  const float* Wo = (const float*)d_in[9];
  const float* bo = (const float*)d_in[10];

  const size_t MD = (size_t)MROWS * D_MODEL;    // 4M elements
  const size_t DD = (size_t)D_MODEL * D_MODEL;  // 1M elements

  char* ws = (char*)d_ws;
  __bf16* qb  = (__bf16*)(ws + 0 * MD * 2);
  __bf16* kb  = (__bf16*)(ws + 1 * MD * 2);
  __bf16* vb  = (__bf16*)(ws + 2 * MD * 2);
  __bf16* WqT = (__bf16*)(ws + 3 * MD * 2 + 0 * DD * 2);
  __bf16* WkT = (__bf16*)(ws + 3 * MD * 2 + 1 * DD * 2);
  __bf16* WvT = (__bf16*)(ws + 3 * MD * 2 + 2 * DD * 2);
  __bf16* WoT = (__bf16*)(ws + 3 * MD * 2 + 3 * DD * 2);
  __bf16* Qh  = (__bf16*)(ws + 3 * MD * 2 + 4 * DD * 2 + 0 * MD * 2);
  __bf16* Kh  = (__bf16*)(ws + 3 * MD * 2 + 4 * DD * 2 + 1 * MD * 2);
  __bf16* Vt  = (__bf16*)(ws + 3 * MD * 2 + 4 * DD * 2 + 2 * MD * 2);
  __bf16* Ctx = (__bf16*)(ws + 3 * MD * 2 + 4 * DD * 2 + 3 * MD * 2);
  // total workspace: 7*MD*2 + 4*DD*2 = 56 MB + 8 MB = 64 MB

  // 1) fp32 -> bf16 activations
  cvt_f32_bf16<<<(int)(MD / 256), 256, 0, stream>>>(q, qb, (int)MD);
  cvt_f32_bf16<<<(int)(MD / 256), 256, 0, stream>>>(k, kb, (int)MD);
  cvt_f32_bf16<<<(int)(MD / 256), 256, 0, stream>>>(v, vb, (int)MD);
  // 2) transpose+convert weights (WT[N][K])
  transpose_w_bf16<<<(int)(DD / 256), 256, 0, stream>>>(Wq, WqT);
  transpose_w_bf16<<<(int)(DD / 256), 256, 0, stream>>>(Wk, WkT);
  transpose_w_bf16<<<(int)(DD / 256), 256, 0, stream>>>(Wv, WvT);
  transpose_w_bf16<<<(int)(DD / 256), 256, 0, stream>>>(Wo, WoT);

  // 3) Q/K projections (row-major) and V projection (transposed head layout)
  dim3 gg(D_MODEL / 128, MROWS / 128);
  gemm_bf16_wmma<0><<<gg, 256, 0, stream>>>(qb, WqT, bq, Qh, MROWS, D_MODEL,
                                            D_MODEL);
  gemm_bf16_wmma<0><<<gg, 256, 0, stream>>>(kb, WkT, bk, Kh, MROWS, D_MODEL,
                                            D_MODEL);
  gemm_bf16_wmma<2><<<gg, 256, 0, stream>>>(vb, WvT, bv, Vt, MROWS, D_MODEL,
                                            D_MODEL);

  // 4) attention
  dim3 ga(SEQ / 128, NHEAD, BATCH);
  flash_attn_wmma<<<ga, 256, 0, stream>>>(Qh, Kh, Vt, Ctx);

  // 5) output projection (f32 out + bias) -> d_out
  gemm_bf16_wmma<1><<<gg, 256, 0, stream>>>(Ctx, WoT, bo, (float*)d_out,
                                            MROWS, D_MODEL, D_MODEL);
}
